// GeometricTopologicalEncoder_16277926052603
// MI455X (gfx1250) — matrix operand order repaired and checked by reading, not compiled, not run on hardware
//
#include <hip/hip_runtime.h>
#include <math.h>

#define B_ 4
#define N_ 256
#define INDIM_ 256
#define E_ 512
#define H_ 8
#define DH_ 64
#define L_ 4
#define ROWS_ (B_*N_)   // 1024

typedef __attribute__((ext_vector_type(16))) _Float16 v16h;
typedef __attribute__((ext_vector_type(8)))  float    v8f;
typedef __attribute__((ext_vector_type(4)))  int      v4i;

#if __has_builtin(__builtin_amdgcn_global_load_async_to_lds_b128)
#define USE_ASYNC_LDS 1
#else
#define USE_ASYNC_LDS 0
#endif

__device__ __forceinline__ void wait_async0() {
#if __has_builtin(__builtin_amdgcn_s_wait_asynccnt)
    __builtin_amdgcn_s_wait_asynccnt(0);
#else
    asm volatile("s_wait_asynccnt 0x0" ::: "memory");
#endif
}

// ---------------------------------------------------------------------------
// Weight convert+transpose: W[K,N] f32 -> Wt[N,K] f16  (once per launch)
// ---------------------------------------------------------------------------
__global__ __launch_bounds__(256)
void convw_kernel(const float* __restrict__ W, _Float16* __restrict__ out,
                  int K, int N)
{
    int idx = blockIdx.x * 256 + threadIdx.x;
    int k = idx / N, n = idx % N;                 // coalesced read
    out[(long)n * K + k] = (_Float16)W[idx];
}

// ---------------------------------------------------------------------------
// Tiled GEMM, WMMA f16 (f32 accumulate):  C = act(A[M,K] @ W + bias)
// A: f32 row-major [M,K].  Wt: f16, pre-transposed [N,K].
// Block = 256 threads = 8 waves (4x2); wave tile 32x32 => 4 WMMAs / K-step.
// Weight tile staged to LDS via async-to-LDS b128 (ASYNCcnt) when available.
// ACT: 0=none 1=silu 2=relu
// ---------------------------------------------------------------------------
#define BM 128
#define BN 64
#define BK 32

template<int ACT>
__global__ __launch_bounds__(256)
void gemm_wmma(const float* __restrict__ A, const _Float16* __restrict__ Wt,
               const float* __restrict__ bias, float* __restrict__ C,
               int M, int N, int K)
{
    __shared__ _Float16 As[BM][BK + 8];   // [m][k]
    __shared__ _Float16 Bs[BN][BK + 8];   // [n][k]  (row pitch 80B, 16B aligned)

    const int tid  = threadIdx.x;
    const int wave = tid >> 5;
    const int lane = tid & 31;
    const int bm = blockIdx.x * BM;
    const int bn = blockIdx.y * BN;
    const int wm = (wave & 3) * 32;
    const int wn = (wave >> 2) * 32;
    const int hl = lane >> 4, lr = lane & 15;

    v8f acc[2][2] = {};

    for (int k0 = 0; k0 < K; k0 += BK) {
        // stage A tile (f32 -> f16), coalesced over k
        for (int i = tid; i < BM * BK; i += 256) {
            int r = i >> 5, c = i & 31;
            As[r][c] = (_Float16)A[(long)(bm + r) * K + (k0 + c)];
        }
        // stage W tile: 64 n-rows x 32 k (64B/row) = 4 x b128 chunks per row
#if USE_ASYNC_LDS
        {
            int n = tid >> 2, chunk = tid & 3;
            const _Float16* gsrc = Wt + (long)(bn + n) * K + k0 + chunk * 8;
            __builtin_amdgcn_global_load_async_to_lds_b128(
                (__attribute__((address_space(1))) v4i*)gsrc,
                (__attribute__((address_space(3))) v4i*)&Bs[n][chunk * 8],
                0, 0);
        }
        wait_async0();
#else
        for (int i = tid; i < BN * BK; i += 256) {
            int n = i >> 5, c = i & 31;
            Bs[n][c] = Wt[(long)(bn + n) * K + (k0 + c)];
        }
#endif
        __syncthreads();

        v16h af[2], bf[2];
        #pragma unroll
        for (int mi = 0; mi < 2; ++mi)
            #pragma unroll
            for (int p = 0; p < 8; ++p) {
                int kk = ((p & 3) * 2) + ((p < 4) ? 0 : 16) + hl * 8;
                af[mi][2*p]   = As[wm + mi*16 + lr][kk];
                af[mi][2*p+1] = As[wm + mi*16 + lr][kk + 1];
            }
        #pragma unroll
        for (int ni = 0; ni < 2; ++ni)
            #pragma unroll
            for (int p = 0; p < 8; ++p) {
                int kk = ((p & 3) * 2) + ((p < 4) ? 0 : 16) + hl * 8;
                bf[ni][2*p]   = Bs[wn + ni*16 + lr][kk];
                bf[ni][2*p+1] = Bs[wn + ni*16 + lr][kk + 1];
            }
        #pragma unroll
        for (int mi = 0; mi < 2; ++mi)
            #pragma unroll
            for (int ni = 0; ni < 2; ++ni)
                acc[mi][ni] = __builtin_amdgcn_wmma_f32_16x16x32_f16(
                    false, af[mi], false, bf[ni], (short)0, acc[mi][ni], false, false);
        __syncthreads();
    }

    #pragma unroll
    for (int mi = 0; mi < 2; ++mi) {
        int row0 = bm + wm + mi*16 + hl * 8;
        #pragma unroll
        for (int ni = 0; ni < 2; ++ni) {
            int col = bn + wn + ni*16 + lr;
            float bv = bias ? bias[col] : 0.f;
            #pragma unroll
            for (int g = 0; g < 8; ++g) {
                float v = acc[mi][ni][g] + bv;
                if (ACT == 1) v = v / (1.f + __expf(-v));
                if (ACT == 2) v = fmaxf(v, 0.f);
                C[(long)(row0 + g) * N + col] = v;
            }
        }
    }
}

// ---------------------------------------------------------------------------
// dist[b,i,j] = safe_norm(coords[b,i]-coords[b,j]); also global max -> dmax
// ---------------------------------------------------------------------------
__global__ __launch_bounds__(256)
void dist_kernel(const float* __restrict__ coords, float* __restrict__ dist,
                 unsigned int* __restrict__ dmax_bits)
{
    __shared__ float red[256];
    int b = blockIdx.x / N_;
    int i = blockIdx.x % N_;
    int j = threadIdx.x;
    float dx = coords[(b*N_ + i)*3 + 0] - coords[(b*N_ + j)*3 + 0];
    float dy = coords[(b*N_ + i)*3 + 1] - coords[(b*N_ + j)*3 + 1];
    float dz = coords[(b*N_ + i)*3 + 2] - coords[(b*N_ + j)*3 + 2];
    float sq = dx*dx + dy*dy + dz*dz;
    float d  = (sq > 0.f) ? sqrtf(sq) : 0.f;
    dist[((long)(b*N_) + i)*N_ + j] = d;
    red[j] = d;
    __syncthreads();
    for (int s = 128; s; s >>= 1) { if (j < s) red[j] = fmaxf(red[j], red[j+s]); __syncthreads(); }
    if (j == 0) atomicMax(dmax_bits, __float_as_uint(red[0]));
}

__global__ void init_dmax(unsigned int* dmax_bits) { *dmax_bits = 0u; }

// ---------------------------------------------------------------------------
// Edge-biased graph attention, one workgroup per (b,i), WMMA score path:
//   A (16x1024 f16): row h<8 = [ 0.125*Q (head-sliced) | G[h] ], rows 8..15 = 0
//   B (1024x256):    [ K^T ; X^T ],  X[j] = relu(ef_ij @ e1 + e1b) staged per
//                    K-tile on the fly.
//   S = A@B  => scores[h][j] = 0.125*Q.K + relu(X).G  (summed in accumulator)
// Then adjacency mask, wave-per-head softmax, VALU att@V.
// ---------------------------------------------------------------------------
__global__ __launch_bounds__(256)
void edge_attn_kernel(const float* __restrict__ Q, const float* __restrict__ Kmat,
                      const float* __restrict__ V,
                      const float* __restrict__ e1w, const float* __restrict__ e1b,
                      const float* __restrict__ e2w,
                      const float* __restrict__ dist, const float* __restrict__ bond,
                      const int* __restrict__ adj, const unsigned int* __restrict__ dmax_bits,
                      float* __restrict__ topo)
{
    __shared__ float Qrow[E_];
    __shared__ float efS[4][N_];
    __shared__ float e1wS[4][E_];
    __shared__ float e1bS[E_];
    __shared__ _Float16 Amat[16][2*E_ + 8];
    __shared__ _Float16 Bs[N_][BK + 8];
    __shared__ float sc[H_][N_];
    __shared__ float hsum[H_];

    const int tid  = threadIdx.x;
    const int wave = tid >> 5;
    const int lane = tid & 31;
    const int hl = lane >> 4, lr = lane & 15;
    const int b = blockIdx.x / N_;
    const int i = blockIdx.x % N_;

    const float* qg = Q + ((long)(b*N_) + i) * E_;
    for (int c = tid; c < E_;   c += 256) Qrow[c] = qg[c];
    for (int c = tid; c < 4*E_; c += 256) e1wS[c >> 9][c & 511] = e1w[c];
    for (int c = tid; c < E_;   c += 256) e1bS[c] = e1b[c];
    {
        int j = tid;
        float dij = dist[((long)(b*N_) + i)*N_ + j];
        float dm  = __uint_as_float(*dmax_bits);
        const float PI = 3.14159265358979f;
        efS[0][j] = bond[((long)(b*N_) + i)*N_ + j];
        efS[1][j] = dij / (dm + 1e-6f);
        efS[2][j] = __cosf(dij * PI);
        efS[3][j] = __sinf(dij * PI);
    }
    __syncthreads();

    // Build A = [maskedScaledQ | G] (f16)
    for (int idx = tid; idx < 16 * 2*E_; idx += 256) {
        int r = idx >> 10;
        int c = idx & (2*E_ - 1);
        float v = 0.f;
        if (r < H_) {
            if (c < E_) {
                v = ((c >> 6) == r) ? Qrow[c] * 0.125f : 0.f;
            } else {
                int cc = c - E_;
                const float* w = e2w + (long)cc * E_ + r * DH_;
                float s = 0.f;
                #pragma unroll 8
                for (int d = 0; d < DH_; ++d) s += Qrow[r*DH_ + d] * w[d];
                v = s;
            }
        }
        Amat[r][c] = (_Float16)v;
    }
    __syncthreads();

    v8f acc0 = {}, acc1 = {};
    const int jb0 = wave * 32;
    for (int k0 = 0; k0 < 2*E_; k0 += BK) {
        for (int idx = tid; idx < N_ * BK; idx += 256) {
            int j = idx >> 5, kk = idx & 31;
            int c = k0 + kk;
            float v;
            if (c < E_) {
                v = Kmat[((long)(b*N_) + j) * E_ + c];
            } else {
                int cc = c - E_;
                float x = e1bS[cc] + efS[0][j]*e1wS[0][cc] + efS[1][j]*e1wS[1][cc]
                                   + efS[2][j]*e1wS[2][cc] + efS[3][j]*e1wS[3][cc];
                v = fmaxf(x, 0.f);
            }
            Bs[j][kk] = (_Float16)v;
        }
        __syncthreads();

        v16h afrag, bfrag0, bfrag1;
        #pragma unroll
        for (int p = 0; p < 8; ++p) {
            int kk = ((p & 3) * 2) + ((p < 4) ? 0 : 16) + hl * 8;
            afrag[2*p]   = Amat[lr][k0 + kk];
            afrag[2*p+1] = Amat[lr][k0 + kk + 1];
        }
        #pragma unroll
        for (int p = 0; p < 8; ++p) {
            int kk = ((p & 3) * 2) + ((p < 4) ? 0 : 16) + hl * 8;
            bfrag0[2*p]   = Bs[jb0 + lr][kk];
            bfrag0[2*p+1] = Bs[jb0 + lr][kk + 1];
            bfrag1[2*p]   = Bs[jb0 + 16 + lr][kk];
            bfrag1[2*p+1] = Bs[jb0 + 16 + lr][kk + 1];
        }
        acc0 = __builtin_amdgcn_wmma_f32_16x16x32_f16(false, afrag, false, bfrag0,
                                                      (short)0, acc0, false, false);
        acc1 = __builtin_amdgcn_wmma_f32_16x16x32_f16(false, afrag, false, bfrag1,
                                                      (short)0, acc1, false, false);
        __syncthreads();
    }

    if (hl == 0) {
        #pragma unroll
        for (int g = 0; g < 8; ++g) {
            sc[g][jb0 + lr]      = acc0[g];
            sc[g][jb0 + 16 + lr] = acc1[g];
        }
    }
    __syncthreads();
    {
        int j = tid;
        if (!adj[((long)(b*N_) + i)*N_ + j]) {
            #pragma unroll
            for (int hh = 0; hh < H_; ++hh) sc[hh][j] = -1e30f;
        }
    }
    __syncthreads();
    {
        const int hh = tid >> 5, l = tid & 31;
        float mx = -1e30f;
        for (int j = l; j < N_; j += 32) mx = fmaxf(mx, sc[hh][j]);
        for (int o = 16; o; o >>= 1) mx = fmaxf(mx, __shfl_xor(mx, o));
        float sm = 0.f;
        for (int j = l; j < N_; j += 32) { float e = __expf(sc[hh][j] - mx); sc[hh][j] = e; sm += e; }
        for (int o = 16; o; o >>= 1) sm += __shfl_xor(sm, o);
        if (l == 0) hsum[hh] = sm;
    }
    __syncthreads();
    #pragma unroll
    for (int rep = 0; rep < 2; ++rep) {
        int c = tid + rep * 256;
        int hh = c >> 6;
        float inv = 1.f / hsum[hh];
        float o = 0.f;
        for (int j = 0; j < N_; ++j) o += sc[hh][j] * V[((long)(b*N_) + j)*E_ + c];
        topo[((long)(b*N_) + i)*E_ + c] = o * inv;
    }
}

// ---------------------------------------------------------------------------
// Plain MHA attention, WMMA score path (A = head-sliced 0.125*Q, 16x512)
// ---------------------------------------------------------------------------
__global__ __launch_bounds__(256)
void mha_attn_kernel(const float* __restrict__ Q, const float* __restrict__ Kmat,
                     const float* __restrict__ V, float* __restrict__ out)
{
    __shared__ float Qrow[E_];
    __shared__ _Float16 Amat[16][E_ + 8];
    __shared__ _Float16 Bs[N_][BK + 8];
    __shared__ float sc[H_][N_];
    __shared__ float hsum[H_];

    const int tid  = threadIdx.x;
    const int wave = tid >> 5;
    const int lane = tid & 31;
    const int hl = lane >> 4, lr = lane & 15;
    const int b = blockIdx.x / N_;
    const int i = blockIdx.x % N_;

    const float* qg = Q + ((long)(b*N_) + i) * E_;
    for (int c = tid; c < E_; c += 256) Qrow[c] = qg[c];
    __syncthreads();
    for (int idx = tid; idx < 16 * E_; idx += 256) {
        int r = idx >> 9, c = idx & 511;
        float v = (r < H_ && (c >> 6) == r) ? Qrow[c] * 0.125f : 0.f;
        Amat[r][c] = (_Float16)v;
    }
    __syncthreads();

    v8f acc0 = {}, acc1 = {};
    const int jb0 = wave * 32;
    for (int k0 = 0; k0 < E_; k0 += BK) {
        for (int idx = tid; idx < N_ * BK; idx += 256) {
            int j = idx >> 5, kk = idx & 31;
            Bs[j][kk] = (_Float16)Kmat[((long)(b*N_) + j) * E_ + (k0 + kk)];
        }
        __syncthreads();
        v16h afrag, bfrag0, bfrag1;
        #pragma unroll
        for (int p = 0; p < 8; ++p) {
            int kk = ((p & 3) * 2) + ((p < 4) ? 0 : 16) + hl * 8;
            afrag[2*p]   = Amat[lr][k0 + kk];
            afrag[2*p+1] = Amat[lr][k0 + kk + 1];
        }
        #pragma unroll
        for (int p = 0; p < 8; ++p) {
            int kk = ((p & 3) * 2) + ((p < 4) ? 0 : 16) + hl * 8;
            bfrag0[2*p]   = Bs[jb0 + lr][kk];
            bfrag0[2*p+1] = Bs[jb0 + lr][kk + 1];
            bfrag1[2*p]   = Bs[jb0 + 16 + lr][kk];
            bfrag1[2*p+1] = Bs[jb0 + 16 + lr][kk + 1];
        }
        acc0 = __builtin_amdgcn_wmma_f32_16x16x32_f16(false, afrag, false, bfrag0,
                                                      (short)0, acc0, false, false);
        acc1 = __builtin_amdgcn_wmma_f32_16x16x32_f16(false, afrag, false, bfrag1,
                                                      (short)0, acc1, false, false);
        __syncthreads();
    }
    if (hl == 0) {
        #pragma unroll
        for (int g = 0; g < 8; ++g) {
            sc[g][jb0 + lr]      = acc0[g];
            sc[g][jb0 + 16 + lr] = acc1[g];
        }
    }
    __syncthreads();
    {
        const int hh = tid >> 5, l = tid & 31;
        float mx = -1e30f;
        for (int j = l; j < N_; j += 32) mx = fmaxf(mx, sc[hh][j]);
        for (int o = 16; o; o >>= 1) mx = fmaxf(mx, __shfl_xor(mx, o));
        float sm = 0.f;
        for (int j = l; j < N_; j += 32) { float e = __expf(sc[hh][j] - mx); sc[hh][j] = e; sm += e; }
        for (int o = 16; o; o >>= 1) sm += __shfl_xor(sm, o);
        if (l == 0) hsum[hh] = sm;
    }
    __syncthreads();
    #pragma unroll
    for (int rep = 0; rep < 2; ++rep) {
        int c = tid + rep * 256;
        int hh = c >> 6;
        float inv = 1.f / hsum[hh];
        float o = 0.f;
        for (int j = 0; j < N_; ++j) o += sc[hh][j] * V[((long)(b*N_) + j)*E_ + c];
        out[((long)(b*N_) + i)*E_ + c] = o * inv;
    }
}

// ---------------------------------------------------------------------------
// out = LayerNorm(x + res) * g + beta   (row = 512; in-place on res allowed)
// ---------------------------------------------------------------------------
__global__ __launch_bounds__(256)
void add_ln_kernel(const float* __restrict__ x, const float* __restrict__ res,
                   const float* __restrict__ g, const float* __restrict__ bta,
                   float* __restrict__ out)
{
    __shared__ float red[256];
    __shared__ float mean_s, rstd_s;
    const int row = blockIdx.x;
    const int tid = threadIdx.x;
    const float* xr = x   + (long)row * E_;
    const float* rr = res + (long)row * E_;
    float v0 = xr[tid]       + rr[tid];
    float v1 = xr[tid + 256] + rr[tid + 256];
    red[tid] = v0 + v1;
    __syncthreads();
    for (int s = 128; s; s >>= 1) { if (tid < s) red[tid] += red[tid+s]; __syncthreads(); }
    if (tid == 0) mean_s = red[0] * (1.f / E_);
    __syncthreads();
    float m = mean_s;
    red[tid] = (v0 - m)*(v0 - m) + (v1 - m)*(v1 - m);
    __syncthreads();
    for (int s = 128; s; s >>= 1) { if (tid < s) red[tid] += red[tid+s]; __syncthreads(); }
    if (tid == 0) rstd_s = rsqrtf(red[0] * (1.f / E_) + 1e-5f);
    __syncthreads();
    float rs = rstd_s;
    out[(long)row*E_ + tid]       = (v0 - m) * rs * g[tid]       + bta[tid];
    out[(long)row*E_ + tid + 256] = (v1 - m) * rs * g[tid + 256] + bta[tid + 256];
}

// ---------------------------------------------------------------------------
// Masked mean pool: out[b,c] = sum_n g[b,n,c]*m / (count + 1e-8)
// ---------------------------------------------------------------------------
__global__ __launch_bounds__(256)
void pool_kernel(const float* __restrict__ gmat, const int* __restrict__ mask,
                 float* __restrict__ out)
{
    const int b = blockIdx.x;
    const int tid = threadIdx.x;
    float cnt = 0.f;
    for (int n = 0; n < N_; ++n) cnt += mask[b*N_ + n] ? 1.f : 0.f;
    float inv = 1.f / (cnt + 1e-8f);
    #pragma unroll
    for (int rep = 0; rep < 2; ++rep) {
        int c = tid + rep * 256;
        float s = 0.f;
        for (int n = 0; n < N_; ++n)
            if (mask[b*N_ + n]) s += gmat[((long)(b*N_) + n)*E_ + c];
        out[(long)b*E_ + c] = s * inv;
    }
}

// ---------------------------------------------------------------------------
// Host-side orchestration
// ---------------------------------------------------------------------------
static void gemm(hipStream_t s, const float* A, const _Float16* Wt, const float* bias,
                 float* C, int M, int N, int K, int act)
{
    dim3 grid(M / BM, N / BN), block(256);
    if (act == 1)      gemm_wmma<1><<<grid, block, 0, s>>>(A, Wt, bias, C, M, N, K);
    else if (act == 2) gemm_wmma<2><<<grid, block, 0, s>>>(A, Wt, bias, C, M, N, K);
    else               gemm_wmma<0><<<grid, block, 0, s>>>(A, Wt, bias, C, M, N, K);
}

extern "C" void kernel_launch(void* const* d_in, const int* in_sizes, int n_in,
                              void* d_out, int out_size, void* d_ws, size_t ws_size,
                              hipStream_t stream)
{
    (void)in_sizes; (void)n_in; (void)out_size; (void)ws_size;

    const float* nf     = (const float*)d_in[0];
    const float* coords = (const float*)d_in[1];
    const int*   adj    = (const int*)  d_in[2];
    const float* bond   = (const float*)d_in[3];
    const int*   amask  = (const int*)  d_in[4];
    // params in dict order (v1/v2/a1/a2 are dead code — never read)
    const float* in_w = (const float*)d_in[5];  const float* in_b = (const float*)d_in[6];
    const float* s1_w = (const float*)d_in[7];  const float* s1_b = (const float*)d_in[8];
    const float* s2_w = (const float*)d_in[9];  const float* s2_b = (const float*)d_in[10];
    const float* q_w  = (const float*)d_in[19]; const float* q_b  = (const float*)d_in[20];
    const float* k_w  = (const float*)d_in[21]; const float* k_b  = (const float*)d_in[22];
    const float* v_w  = (const float*)d_in[23]; const float* v_b  = (const float*)d_in[24];
    const float* e1_w = (const float*)d_in[25]; const float* e1_b = (const float*)d_in[26];
    const float* e2_w = (const float*)d_in[27];
    const float* gln_g= (const float*)d_in[29]; const float* gln_b= (const float*)d_in[30];
    const float* fq_w = (const float*)d_in[31]; const float* fq_b = (const float*)d_in[32];
    const float* fk_w = (const float*)d_in[33]; const float* fk_b = (const float*)d_in[34];
    const float* fv_w = (const float*)d_in[35]; const float* fv_b = (const float*)d_in[36];
    const float* fo_w = (const float*)d_in[37]; const float* fo_b = (const float*)d_in[38];
    const float* ln_g = (const float*)d_in[39]; const float* ln_b = (const float*)d_in[40];
    const float* p1_w = (const float*)d_in[41]; const float* p1_b = (const float*)d_in[42];
    const float* p2_w = (const float*)d_in[43]; const float* p2_b = (const float*)d_in[44];
    const float* gq_w = (const float*)d_in[45]; const float* gq_b = (const float*)d_in[46];
    const float* gk_w = (const float*)d_in[47]; const float* gk_b = (const float*)d_in[48];
    const float* gv_w = (const float*)d_in[49]; const float* gv_b = (const float*)d_in[50];
    const float* go_w = (const float*)d_in[51]; const float* go_b = (const float*)d_in[52];

    float* ws = (float*)d_ws;
    const long MN = (long)ROWS_ * E_;    // 524288 floats
    float* h    = ws;
    float* scal = ws + 1*MN;
    float* qb   = ws + 2*MN;
    float* kb   = ws + 3*MN;
    float* vb   = ws + 4*MN;
    float* topo = ws + 5*MN;
    float* t1   = ws + 6*MN;
    float* fus  = ws + 7*MN;
    float* dist = ws + 8*MN;             // [B,N,N]
    unsigned int* dmax = (unsigned int*)(ws + 8*MN + (long)B_*N_*N_);
    _Float16* whp = (_Float16*)(ws + 8*MN + (long)B_*N_*N_ + 16);  // f16 weight pool

    // --- weight convert + transpose (f32 [K,N] -> f16 [N,K]) -----------------
    auto cw = [&](const float* W, int K, int N) -> const _Float16* {
        _Float16* dst = whp; whp += (long)K * N;
        convw_kernel<<<(K * N) / 256, 256, 0, stream>>>(W, dst, K, N);
        return dst;
    };
    const long EE = (long)E_ * E_;
    const _Float16* in_wt = cw(in_w, INDIM_, E_);
    const _Float16 *s1t[L_], *s2t[L_], *qt[L_], *kt[L_], *vt[L_],
                   *fqt[L_], *fkt[L_], *fvt[L_], *fot[L_];
    for (int i = 0; i < L_; ++i) {
        s1t[i] = cw(s1_w + i*EE, E_, E_);  s2t[i] = cw(s2_w + i*EE, E_, E_);
        qt[i]  = cw(q_w  + i*EE, E_, E_);  kt[i]  = cw(k_w  + i*EE, E_, E_);
        vt[i]  = cw(v_w  + i*EE, E_, E_);
        fqt[i] = cw(fq_w + i*EE, E_, E_);  fkt[i] = cw(fk_w + i*EE, E_, E_);
        fvt[i] = cw(fv_w + i*EE, E_, E_);  fot[i] = cw(fo_w + i*EE, E_, E_);
    }
    const _Float16* p1t = cw(p1_w, E_, E_);
    const _Float16* p2t = cw(p2_w, E_, E_);
    const _Float16* gqt = cw(gq_w, E_, E_);
    const _Float16* gkt = cw(gk_w, E_, E_);
    const _Float16* gvt = cw(gv_w, E_, E_);
    const _Float16* got = cw(go_w, E_, E_);

    // --- geometry -----------------------------------------------------------
    init_dmax<<<1, 1, 0, stream>>>(dmax);
    dist_kernel<<<B_*N_, 256, 0, stream>>>(coords, dist, dmax);

    // --- network ------------------------------------------------------------
    gemm(stream, nf, in_wt, in_b, h, ROWS_, E_, INDIM_, 0);

    for (int i = 0; i < L_; ++i) {
        gemm(stream, h,  s1t[i], s1_b + i*E_, t1,   ROWS_, E_, E_, 1);
        gemm(stream, t1, s2t[i], s2_b + i*E_, scal, ROWS_, E_, E_, 0);

        gemm(stream, h, qt[i], q_b + i*E_, qb, ROWS_, E_, E_, 0);
        gemm(stream, h, kt[i], k_b + i*E_, kb, ROWS_, E_, E_, 0);
        gemm(stream, h, vt[i], v_b + i*E_, vb, ROWS_, E_, E_, 0);

        edge_attn_kernel<<<B_*N_, 256, 0, stream>>>(
            qb, kb, vb, e1_w + (long)i*4*E_, e1_b + i*E_, e2_w + i*EE,
            dist, bond, adj, dmax, topo);
        add_ln_kernel<<<ROWS_, 256, 0, stream>>>(topo, h, gln_g + i*E_, gln_b + i*E_, topo);

        gemm(stream, scal, fqt[i], fq_b + i*E_, qb, ROWS_, E_, E_, 0);
        gemm(stream, topo, fkt[i], fk_b + i*E_, kb, ROWS_, E_, E_, 0);
        gemm(stream, topo, fvt[i], fv_b + i*E_, vb, ROWS_, E_, E_, 0);
        mha_attn_kernel<<<B_*N_, 256, 0, stream>>>(qb, kb, vb, t1);
        gemm(stream, t1, fot[i], fo_b + i*E_, fus, ROWS_, E_, E_, 0);
        add_ln_kernel<<<ROWS_, 256, 0, stream>>>(fus, h, ln_g + i*E_, ln_b + i*E_, h);
    }

    gemm(stream, h,  p1t, p1_b, t1,   ROWS_, E_, E_, 2);
    gemm(stream, t1, p2t, p2_b, scal, ROWS_, E_, E_, 0);
    gemm(stream, scal, gqt, gq_b, qb, ROWS_, E_, E_, 0);
    gemm(stream, scal, gkt, gk_b, kb, ROWS_, E_, E_, 0);
    gemm(stream, scal, gvt, gv_b, vb, ROWS_, E_, E_, 0);
    mha_attn_kernel<<<B_*N_, 256, 0, stream>>>(qb, kb, vb, t1);
    gemm(stream, t1, got, go_b, fus, ROWS_, E_, E_, 0);
    pool_kernel<<<B_, 256, 0, stream>>>(fus, amask, (float*)d_out);
}
